// model_79439715107280
// MI455X (gfx1250) — matrix operand order repaired and checked by reading, not compiled, not run on hardware
//
#include <hip/hip_runtime.h>
#include <hip/hip_bf16.h>

typedef __attribute__((ext_vector_type(16))) _Float16 v16h;
typedef __attribute__((ext_vector_type(8)))  float    v8f;

#define DEV __device__ __forceinline__

namespace {

constexpr int Bsz = 16, Ssz = 256, Tsz = 128, HSZ = 1024, VSZ = 32000;

DEV float sigm(float x) { return 1.f / (1.f + expf(-x)); }

DEV v8f wmma_step(v16h a, v16h b, v8f c) {
  // D = A(16x32 f16) x B(32x16 f16) + C(16x16 f32)
  return __builtin_amdgcn_wmma_f32_16x16x32_f16(false, a, false, b, (short)0, c,
                                                false, false);
}

// B operand: packed as [nTile][kChunk][lane][16 halves]; lane holds column
// n = nTile*16 + (lane&15), halves are K = h + (h>=8 ? 8 : 0) + (lane>=16 ? 8 : 0) within the chunk.
DEV v16h loadB_packed(const _Float16* pk, int nTile, int kcTotal, int kc, int lane) {
  const v16h* p = (const v16h*)(pk + (((size_t)nTile * kcTotal + kc) * 32 + lane) * 16);
  return *p;
}

// A operand assembled from a row-major f16 [16][rowHalves] buffer (LDS or global).
// ISA A layout: VGPR i holds K pair k0 = 2i + (i>=4?8:0) + (lane>=16?8:0).
DEV v16h loadA_rowmajor(const _Float16* buf, int rowHalves, int kc, int lane) {
  int m = lane & 15;
  int loff = (lane >> 4) << 3;
  const unsigned* p32 = (const unsigned*)(buf + m * rowHalves + kc * 32);
  union { v16h v; unsigned u[8]; } r;
#pragma unroll
  for (int i = 0; i < 8; ++i) {
    int k0 = 2 * i + ((i >= 4) ? 8 : 0) + loff;
    r.u[i] = p32[k0 >> 1];
  }
  return r.v;
}

// ---------------- packing kernels ----------------

// w is (N,K) row-major fp32 (used as x @ w.T, so B[k][n] = w[n][k]).
__global__ void k_pack_weight(const float* w, _Float16* dst, int N, int K) {
  long total = (long)N * K;
  int kC = K / 32;
  for (long p = (long)blockIdx.x * blockDim.x + threadIdx.x; p < total;
       p += (long)gridDim.x * blockDim.x) {
    int h    = (int)(p & 15);
    int lane = (int)((p >> 4) & 31);
    long rest = p >> 9;
    int kc = (int)(rest % kC);
    int nT = (int)(rest / kC);
    int k = kc * 32 + h + ((h >= 8) ? 8 : 0) + ((lane >= 16) ? 8 : 0);
    int n = nT * 16 + (lane & 15);
    dst[p] = (_Float16)w[(long)n * K + k];
  }
}

// xs[s] = embE[inp[:, s]] packed as A tiles: [s][kc(32)][lane][16]
__global__ void k_pack_xs(const int* inp, const float* embE, _Float16* dst) {
  const long total = (long)Ssz * 32 * 32 * 16;
  for (long p = (long)blockIdx.x * blockDim.x + threadIdx.x; p < total;
       p += (long)gridDim.x * blockDim.x) {
    int h = (int)(p & 15);
    int lane = (int)((p >> 4) & 31);
    int kc = (int)((p >> 9) & 31);
    int s = (int)(p >> 14);
    int m = lane & 15;
    int k = kc * 32 + h + ((h >= 8) ? 8 : 0) + ((lane >= 16) ? 8 : 0);
    int idx = inp[m * Ssz + s];
    dst[p] = (_Float16)((idx == 0) ? 0.f : embE[(long)idx * HSZ + k]);
  }
}

// ed[t] = embD[prev_seq[t]] packed as A tiles: [t][kc(32)][lane][16]
__global__ void k_pack_ed(const int* outTok, const float* embD, _Float16* dst) {
  const long total = (long)Tsz * 32 * 32 * 16;
  for (long p = (long)blockIdx.x * blockDim.x + threadIdx.x; p < total;
       p += (long)gridDim.x * blockDim.x) {
    int h = (int)(p & 15);
    int lane = (int)((p >> 4) & 31);
    int kc = (int)((p >> 9) & 31);
    int t = (int)(p >> 14);
    int m = lane & 15;
    int k = kc * 32 + h + ((h >= 8) ? 8 : 0) + ((lane >= 16) ? 8 : 0);
    int prev = (t == 0) ? 3 : outTok[m * Tsz + t - 1];
    dst[p] = (_Float16)((prev == 0) ? 0.f : embD[(long)prev * HSZ + k]);
  }
}

// ---------------- parallel pre-gate GEMM: out = A @ B + bias ----------------
// A packed per row: [row][i(aChunks)][lane][16]; out fp32 C-layout [row][nTile][lane][8]
__global__ __launch_bounds__(256) void k_pregate(
    const _Float16* A, const _Float16* Bw, const float* bias, float* outPre,
    int aChunks, int kcStart, int kcTotalB, int nTiles) {
  const int lane = threadIdx.x & 31, wave = threadIdx.x >> 5;
  const int row = blockIdx.x;
  const int nTile = blockIdx.y * 8 + wave;
  const _Float16* Arow = A + (size_t)row * aChunks * 512;
  v8f acc;
#pragma unroll
  for (int v = 0; v < 8; ++v) acc[v] = 0.f;
  for (int i = 0; i < aChunks; ++i) {
    v16h a = *(const v16h*)(Arow + ((size_t)i * 32 + lane) * 16);
    v16h b = loadB_packed(Bw, nTile, kcTotalB, kcStart + i, lane);
    acc = wmma_step(a, b, acc);
  }
  float bv = bias[nTile * 16 + (lane & 15)];
  float* o = outPre + (((size_t)row * nTiles + nTile) * 32 + lane) * 8;
#pragma unroll
  for (int v = 0; v < 8; ++v) o[v] = acc[v] + bv;
}

// ---------------- sequential encoder: block 0 = fwd, block 1 = bwd ----------------
__global__ __launch_bounds__(256) void k_encoder(
    const _Float16* WFhh, const _Float16* WBhh,
    const float* PREF, const float* PREB,
    float* H0, float* C0, _Float16* ENC) {
  __shared__ _Float16 hbuf[16 * 512];   // recurrent h, row-major f16
  const int dir = blockIdx.x;
  const _Float16* Whh = dir ? WBhh : WFhh;
  const float* PRE = dir ? PREB : PREF;
  const int tid = threadIdx.x, lane = tid & 31, wave = tid >> 5;
  const int mLow = lane & 15, mOff = (lane >> 4) << 3;
  const int ht0 = wave * 4;  // 8 waves x 4 hidden tiles = 32 tiles (512 hidden)

  for (int i = tid; i < 16 * 512; i += 256) hbuf[i] = (_Float16)0.f;
  v8f c[4];
#pragma unroll
  for (int ht = 0; ht < 4; ++ht)
#pragma unroll
    for (int v = 0; v < 8; ++v) c[ht][v] = 0.f;
  __syncthreads();

  for (int s = 0; s < Ssz; ++s) {
    const int rowPre = dir ? (Ssz - 1 - s) : s;
    v8f acc[4][4];
#pragma unroll
    for (int ht = 0; ht < 4; ++ht)
#pragma unroll
      for (int g = 0; g < 4; ++g) {
        const float* pp =
            PRE + (((size_t)rowPre * 128 + (g * 32 + ht0 + ht)) * 32 + lane) * 8;
#pragma unroll
        for (int v = 0; v < 8; ++v) acc[ht][g][v] = pp[v];
      }
    for (int kc = 0; kc < 16; ++kc) {   // K = 512 recurrent
      v16h a = loadA_rowmajor(hbuf, 512, kc, lane);
#pragma unroll
      for (int ht = 0; ht < 4; ++ht)
#pragma unroll
        for (int g = 0; g < 4; ++g) {
          v16h b = loadB_packed(Whh, g * 32 + ht0 + ht, 16, kc, lane);
          acc[ht][g] = wmma_step(a, b, acc[ht][g]);
        }
    }
    __syncthreads();
#pragma unroll
    for (int ht = 0; ht < 4; ++ht) {
      int n = (ht0 + ht) * 16 + mLow;
#pragma unroll
      for (int v = 0; v < 8; ++v) {
        float i_ = sigm(acc[ht][0][v]);
        float f_ = sigm(acc[ht][1][v]);
        float g_ = tanhf(acc[ht][2][v]);
        float o_ = sigm(acc[ht][3][v]);
        float cn = f_ * c[ht][v] + i_ * g_;
        c[ht][v] = cn;
        float hv = o_ * tanhf(cn);
        int m = v + mOff;
        hbuf[m * 512 + n] = (_Float16)hv;
        int sStore = dir ? (Ssz - 1 - s) : s;
        ENC[((size_t)m * Ssz + sStore) * 1024 + dir * 512 + n] = (_Float16)hv;
        if (s == Ssz - 1) {
          H0[m * 1024 + dir * 512 + n] = hv;
          C0[m * 1024 + dir * 512 + n] = cn;
        }
      }
    }
    __syncthreads();
  }
}

// ---------------- sequential decoder: 1 block, 32 waves ----------------
__global__ __launch_bounds__(1024) void k_decoder(
    const int* outTok,
    const _Float16* WIih, const _Float16* WIhh, const float* int_b,
    const _Float16* WDih, const _Float16* WDhh,
    const _Float16* Wli, const _Float16* Wlo,
    const float* PRED, const float* H0, const float* C0,
    const _Float16* ENC, _Float16* OPG) {
  extern __shared__ char smem[];
  _Float16* hL  = (_Float16*)smem;          // dec h     16x1024 f16
  _Float16* hiL = hL + 16 * 1024;           // int hi
  _Float16* ioL = hiL + 16 * 1024;          // iout
  _Float16* opL = ioL + 16 * 1024;          // op
  _Float16* ccL = opL + 16 * 1024;          // attention context
  float* qL  = (float*)(ccL + 16 * 1024);   // q 16x1024 f32
  float* scL = qL + 16 * 1024;              // scores 16x256 f32

  const int tid = threadIdx.x, lane = tid & 31, wave = tid >> 5;
  const int mLow = lane & 15, mOff = (lane >> 4) << 3;

  for (int i = tid; i < 16 * 1024; i += 1024) {
    float v = H0[i];
    hL[i] = (_Float16)v;
    hiL[i] = (_Float16)v;
    opL[i] = (_Float16)0.f;
    ioL[i] = (_Float16)0.f;
  }
  v8f ci[2], cd[2];   // cell states resident in VGPRs (C-layout), owned per wave
#pragma unroll
  for (int j = 0; j < 2; ++j) {
    int n = (wave * 2 + j) * 16 + mLow;
#pragma unroll
    for (int v = 0; v < 8; ++v) {
      int m = v + mOff;
      ci[j][v] = C0[m * 1024 + n];
      cd[j][v] = C0[m * 1024 + n];
    }
  }
  __syncthreads();

  // preamble: (hi, ci, iout) = lstm_step(0, h0, c0) on the intermediate LSTM
  {
    v8f acc[2][4];
#pragma unroll
    for (int j = 0; j < 2; ++j)
#pragma unroll
      for (int g = 0; g < 4; ++g) {
        float bv = int_b[(g * 64 + wave * 2 + j) * 16 + mLow];
#pragma unroll
        for (int v = 0; v < 8; ++v) acc[j][g][v] = bv;
      }
    for (int kc = 0; kc < 32; ++kc) {
      v16h a = loadA_rowmajor(hiL, 1024, kc, lane);
#pragma unroll
      for (int j = 0; j < 2; ++j)
#pragma unroll
        for (int g = 0; g < 4; ++g) {
          v16h b = loadB_packed(WIhh, g * 64 + wave * 2 + j, 32, kc, lane);
          acc[j][g] = wmma_step(a, b, acc[j][g]);
        }
    }
    v8f hnew[2];
#pragma unroll
    for (int j = 0; j < 2; ++j)
#pragma unroll
      for (int v = 0; v < 8; ++v) {
        float i_ = sigm(acc[j][0][v]);
        float f_ = sigm(acc[j][1][v]);
        float g_ = tanhf(acc[j][2][v]);
        float o_ = sigm(acc[j][3][v]);
        float cn = f_ * ci[j][v] + i_ * g_;
        ci[j][v] = cn;
        hnew[j][v] = o_ * tanhf(cn);
      }
    __syncthreads();
#pragma unroll
    for (int j = 0; j < 2; ++j) {
      int n = (wave * 2 + j) * 16 + mLow;
#pragma unroll
      for (int v = 0; v < 8; ++v) {
        int m = v + mOff;
        hiL[m * 1024 + n] = (_Float16)hnew[j][v];
        ioL[m * 1024 + n] = (_Float16)hnew[j][v];
      }
    }
    __syncthreads();
  }

  for (int t = 0; t < Tsz; ++t) {
    float msel[8];
#pragma unroll
    for (int v = 0; v < 8; ++v) {
      int m = v + mOff;
      if (t == 0) msel[v] = 0.f;
      else {
        int prev = outTok[m * Tsz + t - 1];
        msel[v] = (prev == 4 || prev == 5) ? 1.f : 0.f;
      }
    }

    // (a,b) intermediate LSTM: gates = [op, iout]@Wih + hi@Whh + b, masked update
    {
      v8f acc[2][4];
#pragma unroll
      for (int j = 0; j < 2; ++j)
#pragma unroll
        for (int g = 0; g < 4; ++g) {
          float bv = int_b[(g * 64 + wave * 2 + j) * 16 + mLow];
#pragma unroll
          for (int v = 0; v < 8; ++v) acc[j][g][v] = bv;
        }
      for (int kc = 0; kc < 64; ++kc) {
        const _Float16* srcA = (kc < 32) ? opL : ioL;
        v16h a = loadA_rowmajor(srcA, 1024, kc & 31, lane);
#pragma unroll
        for (int j = 0; j < 2; ++j)
#pragma unroll
          for (int g = 0; g < 4; ++g) {
            v16h b = loadB_packed(WIih, g * 64 + wave * 2 + j, 64, kc, lane);
            acc[j][g] = wmma_step(a, b, acc[j][g]);
          }
      }
      for (int kc = 0; kc < 32; ++kc) {
        v16h a = loadA_rowmajor(hiL, 1024, kc, lane);
#pragma unroll
        for (int j = 0; j < 2; ++j)
#pragma unroll
          for (int g = 0; g < 4; ++g) {
            v16h b = loadB_packed(WIhh, g * 64 + wave * 2 + j, 32, kc, lane);
            acc[j][g] = wmma_step(a, b, acc[j][g]);
          }
      }
      v8f hi2[2], ci2[2];
#pragma unroll
      for (int j = 0; j < 2; ++j)
#pragma unroll
        for (int v = 0; v < 8; ++v) {
          float i_ = sigm(acc[j][0][v]);
          float f_ = sigm(acc[j][1][v]);
          float g_ = tanhf(acc[j][2][v]);
          float o_ = sigm(acc[j][3][v]);
          float cn = f_ * ci[j][v] + i_ * g_;
          ci2[j][v] = cn;
          hi2[j][v] = o_ * tanhf(cn);
        }
      __syncthreads();
#pragma unroll
      for (int j = 0; j < 2; ++j) {
        int n = (wave * 2 + j) * 16 + mLow;
#pragma unroll
        for (int v = 0; v < 8; ++v) {
          int m = v + mOff;
          bool on = msel[v] > 0.5f;
          float hiOld = (float)hiL[m * 1024 + n];
          float ioOld = (float)ioL[m * 1024 + n];
          float hiNew = on ? hi2[j][v] : hiOld;
          float ioNew = on ? hi2[j][v] : ioOld;
          ci[j][v] = on ? ci2[j][v] : ci[j][v];
          hiL[m * 1024 + n] = (_Float16)hiNew;
          ioL[m * 1024 + n] = (_Float16)ioNew;
        }
      }
      __syncthreads();
    }

    // (c) decoder LSTM: gates = PRED(emb part + bias) + [op, iout]@Wih[1024:] + h@Whh
    {
      v8f acc[2][4];
#pragma unroll
      for (int j = 0; j < 2; ++j)
#pragma unroll
        for (int g = 0; g < 4; ++g) {
          const float* pp =
              PRED + (((size_t)t * 256 + (g * 64 + wave * 2 + j)) * 32 + lane) * 8;
#pragma unroll
          for (int v = 0; v < 8; ++v) acc[j][g][v] = pp[v];
        }
      for (int kc = 32; kc < 96; ++kc) {
        const _Float16* srcA = (kc < 64) ? opL : ioL;
        v16h a = loadA_rowmajor(srcA, 1024, kc & 31, lane);
#pragma unroll
        for (int j = 0; j < 2; ++j)
#pragma unroll
          for (int g = 0; g < 4; ++g) {
            v16h b = loadB_packed(WDih, g * 64 + wave * 2 + j, 96, kc, lane);
            acc[j][g] = wmma_step(a, b, acc[j][g]);
          }
      }
      for (int kc = 0; kc < 32; ++kc) {
        v16h a = loadA_rowmajor(hL, 1024, kc, lane);
#pragma unroll
        for (int j = 0; j < 2; ++j)
#pragma unroll
          for (int g = 0; g < 4; ++g) {
            v16h b = loadB_packed(WDhh, g * 64 + wave * 2 + j, 32, kc, lane);
            acc[j][g] = wmma_step(a, b, acc[j][g]);
          }
      }
      v8f hnew[2];
#pragma unroll
      for (int j = 0; j < 2; ++j)
#pragma unroll
        for (int v = 0; v < 8; ++v) {
          float i_ = sigm(acc[j][0][v]);
          float f_ = sigm(acc[j][1][v]);
          float g_ = tanhf(acc[j][2][v]);
          float o_ = sigm(acc[j][3][v]);
          float cn = f_ * cd[j][v] + i_ * g_;
          cd[j][v] = cn;
          hnew[j][v] = o_ * tanhf(cn);
        }
      __syncthreads();
#pragma unroll
      for (int j = 0; j < 2; ++j) {
        int n = (wave * 2 + j) * 16 + mLow;
#pragma unroll
        for (int v = 0; v < 8; ++v)
          hL[(v + mOff) * 1024 + n] = (_Float16)hnew[j][v];
      }
      __syncthreads();
    }

    // (d) q = h @ linin.T
    {
      v8f acc2[2];
#pragma unroll
      for (int j = 0; j < 2; ++j)
#pragma unroll
        for (int v = 0; v < 8; ++v) acc2[j][v] = 0.f;
      for (int kc = 0; kc < 32; ++kc) {
        v16h a = loadA_rowmajor(hL, 1024, kc, lane);
#pragma unroll
        for (int j = 0; j < 2; ++j) {
          v16h b = loadB_packed(Wli, wave * 2 + j, 32, kc, lane);
          acc2[j] = wmma_step(a, b, acc2[j]);
        }
      }
#pragma unroll
      for (int j = 0; j < 2; ++j) {
        int n = (wave * 2 + j) * 16 + mLow;
#pragma unroll
        for (int v = 0; v < 8; ++v) qL[(v + mOff) * 1024 + n] = acc2[j][v];
      }
      __syncthreads();
    }

    // (e) attention scores
#pragma unroll
    for (int i = 0; i < 4; ++i) {
      int p = tid + i * 1024;
      int b = p >> 8, s = p & 255;
      const _Float16* e = ENC + ((size_t)b * Ssz + s) * 1024;
      const float* q = qL + b * 1024;
      float sum = 0.f;
      for (int h = 0; h < 1024; ++h) sum += (float)e[h] * q[h];
      scL[b * 256 + s] = sum;
    }
    __syncthreads();

    // (f) softmax over s (wave w handles batch b=w, wave32 shuffle reduction)
    if (wave < 16) {
      int b = wave;
      float mx = -3.4e38f;
#pragma unroll
      for (int i = 0; i < 8; ++i) mx = fmaxf(mx, scL[b * 256 + lane + i * 32]);
      for (int off = 16; off > 0; off >>= 1) mx = fmaxf(mx, __shfl_xor(mx, off));
      float sm = 0.f;
#pragma unroll
      for (int i = 0; i < 8; ++i) {
        int idx = b * 256 + lane + i * 32;
        float eo = expf(scL[idx] - mx);
        scL[idx] = eo;
        sm += eo;
      }
      for (int off = 16; off > 0; off >>= 1) sm += __shfl_xor(sm, off);
      float inv = 1.f / sm;
#pragma unroll
      for (int i = 0; i < 8; ++i) scL[b * 256 + lane + i * 32] *= inv;
    }
    __syncthreads();

    // (g) context cc = sum_s w[s] * enc[b,s,:]
#pragma unroll
    for (int i = 0; i < 16; ++i) {
      int o = tid + i * 1024;
      int b = o >> 10, col = o & 1023;
      const _Float16* e = ENC + ((size_t)b * Ssz) * 1024 + col;
      const float* w = scL + b * 256;
      float sum = 0.f;
      for (int s = 0; s < 256; ++s) sum += w[s] * (float)e[(size_t)s * 1024];
      ccL[b * 1024 + col] = (_Float16)sum;
    }
    __syncthreads();

    // (h) op = tanh([cc, h] @ linout.T); store for the final generator GEMM
    {
      v8f acc2[2];
#pragma unroll
      for (int j = 0; j < 2; ++j)
#pragma unroll
        for (int v = 0; v < 8; ++v) acc2[j][v] = 0.f;
      for (int kc = 0; kc < 64; ++kc) {
        const _Float16* srcA = (kc < 32) ? ccL : hL;
        v16h a = loadA_rowmajor(srcA, 1024, kc & 31, lane);
#pragma unroll
        for (int j = 0; j < 2; ++j) {
          v16h b = loadB_packed(Wlo, wave * 2 + j, 64, kc, lane);
          acc2[j] = wmma_step(a, b, acc2[j]);
        }
      }
      __syncthreads();
#pragma unroll
      for (int j = 0; j < 2; ++j) {
        int n = (wave * 2 + j) * 16 + mLow;
#pragma unroll
        for (int v = 0; v < 8; ++v) {
          int m = v + mOff;
          float ov = tanhf(acc2[j][v]);
          opL[m * 1024 + n] = (_Float16)ov;
          OPG[(size_t)t * 16 * 1024 + m * 1024 + n] = (_Float16)ov;
        }
      }
      __syncthreads();
    }
  }
}

// ---------------- final generator GEMM: logits[b,t,:] = op[t] @ gen_w.T + gen_b ----------------
__global__ __launch_bounds__(256) void k_gen(const _Float16* OPG, const _Float16* WG,
                                             const float* gen_b, float* outp) {
  __shared__ _Float16 aop[16 * 1024];
  const int t = blockIdx.y;
  const unsigned* src = (const unsigned*)(OPG + (size_t)t * 16 * 1024);
  unsigned* dst = (unsigned*)aop;
  for (int i = threadIdx.x; i < 16 * 1024 / 2; i += 256) dst[i] = src[i];
  __syncthreads();
  const int lane = threadIdx.x & 31, wave = threadIdx.x >> 5;
  const int mLow = lane & 15, mOff = (lane >> 4) << 3;
  const int nTile = blockIdx.x * 8 + wave;
  v8f acc;
#pragma unroll
  for (int v = 0; v < 8; ++v) acc[v] = 0.f;
  for (int kc = 0; kc < 32; ++kc) {
    v16h a = loadA_rowmajor(aop, 1024, kc, lane);
    v16h b = loadB_packed(WG, nTile, 32, kc, lane);
    acc = wmma_step(a, b, acc);
  }
  float bv = gen_b[nTile * 16 + mLow];
#pragma unroll
  for (int v = 0; v < 8; ++v) {
    int m = v + mOff;
    outp[((size_t)m * Tsz + t) * VSZ + nTile * 16 + mLow] = acc[v] + bv;
  }
}

}  // namespace

extern "C" void kernel_launch(void* const* d_in, const int* in_sizes, int n_in,
                              void* d_out, int out_size, void* d_ws, size_t ws_size,
                              hipStream_t stream) {
  (void)in_sizes; (void)n_in; (void)out_size; (void)ws_size;
  const int*   inp   = (const int*)d_in[0];
  const int*   outk  = (const int*)d_in[1];
  const float* embE  = (const float*)d_in[2];
  const float* embD  = (const float*)d_in[3];
  const float* wfih  = (const float*)d_in[4];
  const float* wfhh  = (const float*)d_in[5];
  const float* bf    = (const float*)d_in[6];
  const float* wbih  = (const float*)d_in[7];
  const float* wbhh  = (const float*)d_in[8];
  const float* bbk   = (const float*)d_in[9];
  const float* wiih  = (const float*)d_in[10];
  const float* wihh  = (const float*)d_in[11];
  const float* bi    = (const float*)d_in[12];
  const float* wdih  = (const float*)d_in[13];
  const float* wdhh  = (const float*)d_in[14];
  const float* bd    = (const float*)d_in[15];
  const float* wli   = (const float*)d_in[16];
  const float* wlo   = (const float*)d_in[17];
  const float* wg    = (const float*)d_in[18];
  const float* gb    = (const float*)d_in[19];
  float* logits = (float*)d_out;

  char* ws = (char*)d_ws;
  size_t cur = 0;
  auto take = [&](size_t bytes) -> char* {
    char* p = ws + cur;
    cur += (bytes + 255) & ~(size_t)255;
    return p;
  };
  _Float16* WFih = (_Float16*)take((size_t)2048 * 1024 * 2);
  _Float16* WFhh = (_Float16*)take((size_t)2048 * 512 * 2);
  _Float16* WBih = (_Float16*)take((size_t)2048 * 1024 * 2);
  _Float16* WBhh = (_Float16*)take((size_t)2048 * 512 * 2);
  _Float16* WIih = (_Float16*)take((size_t)4096 * 2048 * 2);
  _Float16* WIhh = (_Float16*)take((size_t)4096 * 1024 * 2);
  _Float16* WDih = (_Float16*)take((size_t)4096 * 3072 * 2);
  _Float16* WDhh = (_Float16*)take((size_t)4096 * 1024 * 2);
  _Float16* WLi  = (_Float16*)take((size_t)1024 * 1024 * 2);
  _Float16* WLo  = (_Float16*)take((size_t)1024 * 2048 * 2);
  _Float16* WG   = (_Float16*)take((size_t)32000 * 1024 * 2);
  _Float16* XS   = (_Float16*)take((size_t)256 * 32 * 512 * 2);
  _Float16* ED   = (_Float16*)take((size_t)128 * 32 * 512 * 2);
  float* PREF = (float*)take((size_t)256 * 128 * 256 * 4);
  float* PREB = (float*)take((size_t)256 * 128 * 256 * 4);
  float* PRED = (float*)take((size_t)128 * 256 * 256 * 4);
  float* H0 = (float*)take((size_t)16 * 1024 * 4);
  float* C0 = (float*)take((size_t)16 * 1024 * 4);
  _Float16* ENC = (_Float16*)take((size_t)16 * 256 * 1024 * 2);
  _Float16* OPG = (_Float16*)take((size_t)128 * 16 * 1024 * 2);

  auto packW = [&](const float* w, _Float16* dst, int N, int K) {
    long total = (long)N * K;
    int blocks = (int)((total + 255) / 256);
    if (blocks > 8192) blocks = 8192;
    k_pack_weight<<<blocks, 256, 0, stream>>>(w, dst, N, K);
  };
  packW(wfih, WFih, 2048, 1024);
  packW(wfhh, WFhh, 2048, 512);
  packW(wbih, WBih, 2048, 1024);
  packW(wbhh, WBhh, 2048, 512);
  packW(wiih, WIih, 4096, 2048);
  packW(wihh, WIhh, 4096, 1024);
  packW(wdih, WDih, 4096, 3072);
  packW(wdhh, WDhh, 4096, 1024);
  packW(wli,  WLi,  1024, 1024);
  packW(wlo,  WLo,  1024, 2048);
  packW(wg,   WG,   32000, 1024);

  k_pack_xs<<<8192, 256, 0, stream>>>(inp, embE, XS);
  k_pack_ed<<<4096, 256, 0, stream>>>(outk, embD, ED);

  // parallel input projections (bias folded in)
  k_pregate<<<dim3(256, 16), 256, 0, stream>>>(XS, WFih, bf,  PREF, 32, 0, 32, 128);
  k_pregate<<<dim3(256, 16), 256, 0, stream>>>(XS, WBih, bbk, PREB, 32, 0, 32, 128);
  k_pregate<<<dim3(128, 32), 256, 0, stream>>>(ED, WDih, bd,  PRED, 32, 0, 96, 256);

  k_encoder<<<2, 256, 0, stream>>>(WFhh, WBhh, PREF, PREB, H0, C0, ENC);

  size_t dynLds = (size_t)5 * 16 * 1024 * 2 + (size_t)16 * 1024 * 4 + (size_t)16 * 256 * 4;
  k_decoder<<<1, 1024, dynLds, stream>>>(outk, WIih, WIhh, bi, WDih, WDhh, WLi, WLo,
                                         PRED, H0, C0, ENC, OPG);

  k_gen<<<dim3(250, 128), 256, 0, stream>>>(OPG, WG, gb, logits);
}
